// SoftModuleVAE_3221225472661
// MI455X (gfx1250) — matrix operand order repaired and checked by reading, not compiled, not run on hardware
//
#include <hip/hip_runtime.h>
#include <hip/hip_bf16.h>

// SoftModuleVAE forward for MI455X (gfx1250, wave32).
// All dense layers run through v_wmma_f32_16x16x32_bf16 (f32 accumulate),
// with fused bias / relu / elementwise-e-multiply / bf16-store epilogues.
//
// Input pointer order (jax tree-flatten of setup_inputs(), dict keys sorted):
//  0 x(4096x512) 1 emb_in(4096x64) 2 eps(4096x128)
//  base: 3 W0(512x400) 4 b0 5 W1(400x400) 6 b1
//  de_base: 7 W(128x400) 8 b
//  dec_gate: 9 Wc0(64x512) 10 bc0 11 Wc1(128x512) 12 bc1 13 Wcl(192x512) 14 bcl
//            15 F0W(400x512) 16 F0b 17 F1W(512x512) 18 F1b 19 w0W(512x64) 20 w0b
//            21 Wf0(512x64) 22 bf0 23 Wf1(512x64) 24 bf1 25 WlW(512x8) 26 wlb
//  dec_last: 27 W(400x512) 28 b
//  dec_modules: 29 W(4x8x400x400) 30 b 31 b1 32 b2 33 g1 34 g2
//  em_base: 35 W(64x400) 36 b
//  enc_gate: 37..54 (same layout as dec_gate)
//  enc_last: 55 W(400x256) 56 b
//  enc_modules: 57 W 58 b 59 b1 60 b2 61 g1 62 g2

typedef unsigned short u16;
typedef __attribute__((ext_vector_type(16))) __bf16 v16bf;
typedef __attribute__((ext_vector_type(8)))  float  v8f;
typedef __attribute__((ext_vector_type(4)))  unsigned int u32x4;

__device__ __forceinline__ u16 f2bf(float f) {
  unsigned int u = __float_as_uint(f);
  u += 0x7fffu + ((u >> 16) & 1u);        // round-to-nearest-even
  return (u16)(u >> 16);
}

union Frag { u32x4 q[2]; v16bf v; };

// ---------------------------------------------------------------------------
// WMMA GEMM: C[4096 x N] = act( A[4096 x K](bf16) @ B[K x N] + bias )
// B pre-transposed/padded to (rowsAlloc x Kpad) bf16, zero-filled.
// FLAGS: 1=RELU 2=EMUL(out*=emul[row,col]) 4=OUT_BF16 8=BIAS
// ---------------------------------------------------------------------------
template <int FLAGS>
__global__ __launch_bounds__(256) void k_gemm(
    const u16* __restrict__ A, int lda, long aB,
    const u16* __restrict__ BT, int ldb, long bB,
    const float* __restrict__ bias, long biasB,
    const float* __restrict__ emul, int lde,
    float* __restrict__ Cf, u16* __restrict__ Cb, int ldc, long cB,
    int N, int nPad, int ksteps) {
  const int lane = threadIdx.x & 31, wave = threadIdx.x >> 5;
  const int wr = wave & 3, wc = wave >> 2;
  const int tm0 = blockIdx.x * 128, tn0 = blockIdx.y * 128, z = blockIdx.z;
  A  += (long)z * aB;
  BT += (long)z * bB;
  const int ml = lane & 15, hf = lane >> 4;

  const u16* Ar[2];
  Ar[0] = A + (long)(tm0 + wr * 32 + ml) * lda + hf * 8;
  Ar[1] = Ar[0] + (long)16 * lda;
  const u16* Bc[4];
#pragma unroll
  for (int c = 0; c < 4; ++c)
    Bc[c] = BT + (long)(tn0 + wc * 64 + c * 16 + ml) * ldb + hf * 8;

  v8f acc[2][4];
  const v8f vz = {};
#pragma unroll
  for (int r = 0; r < 2; ++r)
#pragma unroll
    for (int c = 0; c < 4; ++c) acc[r][c] = vz;

  for (int ks = 0; ks < ksteps; ++ks) {
    const int k0 = ks * 32;
    Frag a[2], b[4];
#pragma unroll
    for (int r = 0; r < 2; ++r) {
      a[r].q[0] = *(const u32x4*)(Ar[r] + k0);
      a[r].q[1] = *(const u32x4*)(Ar[r] + k0 + 16);
    }
#pragma unroll
    for (int c = 0; c < 4; ++c) {
      b[c].q[0] = *(const u32x4*)(Bc[c] + k0);
      b[c].q[1] = *(const u32x4*)(Bc[c] + k0 + 16);
    }
    if (ks + 1 < ksteps) {                 // global_prefetch_b8 next K-slab
      __builtin_prefetch(Ar[0] + k0 + 32, 0, 0);
      __builtin_prefetch(Bc[0] + k0 + 32, 0, 0);
    }
#pragma unroll
    for (int r = 0; r < 2; ++r)
#pragma unroll
      for (int c = 0; c < 4; ++c)
        acc[r][c] = __builtin_amdgcn_wmma_f32_16x16x32_bf16(
            false, a[r].v, false, b[c].v, (short)0, acc[r][c], false, false);
  }

  const int rb = hf << 3;
  const float* biasz = (FLAGS & 8) ? bias + (long)z * biasB : nullptr;
  const long cz = (long)z * cB;
#pragma unroll
  for (int r = 0; r < 2; ++r) {
#pragma unroll
    for (int c = 0; c < 4; ++c) {
      const int col = tn0 + wc * 64 + c * 16 + ml;
      if (col >= nPad) continue;
      const bool valid = col < N;
      const float bv = ((FLAGS & 8) && valid) ? biasz[col] : 0.f;
#pragma unroll
      for (int v = 0; v < 8; ++v) {
        const int row = tm0 + wr * 32 + r * 16 + rb + v;
        float val = 0.f;
        if (valid) {
          val = acc[r][c][v] + bv;
          if (FLAGS & 2) val *= emul[(long)row * lde + col];
          if (FLAGS & 1) val = fmaxf(val, 0.f);
        }
        if (FLAGS & 4) Cb[cz + (long)row * ldc + col] = f2bf(val);
        else           Cf[cz + (long)row * ldc + col] = val;
      }
    }
  }
}

// ---------------------------------------------------------------------------
// Support kernels
// ---------------------------------------------------------------------------
__global__ void k_prepw(const float* __restrict__ W, u16* __restrict__ dst,
                        int K, int N, int Kp, long total) {
  long i = (long)blockIdx.x * blockDim.x + threadIdx.x;
  if (i >= total) return;
  int k = (int)(i % Kp);
  long n = i / Kp;
  float v = (n < N && k < K) ? W[(long)k * N + n] : 0.f;
  dst[i] = f2bf(v);
}

__global__ void k_cvt(const float* __restrict__ s, u16* __restrict__ d,
                      int N, int stride, int relu, long total) {
  long i = (long)blockIdx.x * blockDim.x + threadIdx.x;
  if (i >= total) return;
  int col = (int)(i % stride);
  long row = i / stride;
  float v = 0.f;
  if (col < N) { v = s[row * N + col]; if (relu) v = fmaxf(v, 0.f); }
  d[i] = f2bf(v);
}

__global__ void k_mulrelu(const float* __restrict__ a, const float* __restrict__ b,
                          u16* __restrict__ d, int N, int stride, long total) {
  long i = (long)blockIdx.x * blockDim.x + threadIdx.x;
  if (i >= total) return;
  int col = (int)(i % stride);
  long row = i / stride;
  float v = 0.f;
  if (col < N) v = fmaxf(a[row * N + col] * b[row * N + col], 0.f);
  d[i] = f2bf(v);
}

__global__ void k_colstats(const float* __restrict__ x, int rows, int C,
                           float* __restrict__ mean, float* __restrict__ rstd) {
  __shared__ float ss[8][32], sq[8][32];
  int tx = threadIdx.x & 31, ty = threadIdx.x >> 5;
  int c = blockIdx.x * 32 + tx;
  float s = 0.f, q = 0.f;
  for (int r = ty; r < rows; r += 8) {
    float v = x[(long)r * C + c];
    s += v; q += v * v;
  }
  ss[ty][tx] = s; sq[ty][tx] = q;
  __syncthreads();
  if (ty == 0) {
    for (int k = 1; k < 8; ++k) { s += ss[k][tx]; q += sq[k][tx]; }
    float m = s / rows, var = q / rows - m * m;
    mean[c] = m;
    rstd[c] = rsqrtf(var + 1e-5f);
  }
}

__global__ void k_bn_bf(const float* __restrict__ x, const float* __restrict__ mean,
                        const float* __restrict__ rstd, const float* __restrict__ g,
                        const float* __restrict__ be, u16* __restrict__ dst, long total) {
  long i = (long)blockIdx.x * blockDim.x + threadIdx.x;
  if (i >= total) return;
  int hp = (int)(i % 416);
  long bm = i / 416;
  int m = (int)(bm & 7);
  long b = bm >> 3;
  u16 o = 0;
  if (hp < 400) {
    int c = m * 400 + hp;
    o = f2bf((x[b * 3200 + c] - mean[c]) * rstd[c] * g[c] + be[c]);
  }
  dst[i] = o;
}

__global__ void k_bn_f32(const float* __restrict__ x, const float* __restrict__ mean,
                         const float* __restrict__ rstd, const float* __restrict__ g,
                         const float* __restrict__ be, float* __restrict__ dst, long total) {
  long i = (long)blockIdx.x * blockDim.x + threadIdx.x;
  if (i >= total) return;
  int c = (int)(i % 3200);
  dst[i] = (x[i] - mean[c]) * rstd[c] * g[c] + be[c];
}

__global__ void k_bcast_relu(const float* __restrict__ src, float* __restrict__ dst, long total) {
  long i = (long)blockIdx.x * blockDim.x + threadIdx.x;
  if (i >= total) return;
  int c = (int)(i % 3200);
  int h = c % 400;
  long b = i / 3200;
  dst[i] = fmaxf(src[b * 400 + h], 0.f);
}

__global__ void k_mixture(const float* __restrict__ x, const float* __restrict__ wf,
                          int woff, float* __restrict__ out, long total) {
  long i = (long)blockIdx.x * blockDim.x + threadIdx.x;
  if (i >= total) return;
  int h = (int)(i % 400);
  long b = i / 400;
  const float* xb = x + b * 3200 + h;
  float xv[8];
#pragma unroll
  for (int k = 0; k < 8; ++k) xv[k] = xb[k * 400];
  const float* wb = wf + b * 192 + woff;
#pragma unroll
  for (int j = 0; j < 8; ++j) {
    float s = 0.f;
#pragma unroll
    for (int k = 0; k < 8; ++k) s += wb[j * 8 + k] * xv[k];
    out[(b * 8 + j) * 400 + h] = fmaxf(s, 0.f);
  }
}

__global__ void k_softmax8(const float* __restrict__ raw, int G, long total,
                           float* __restrict__ dF, int fs, int fo,
                           u16* __restrict__ dB, int bs, int bo) {
  long i = (long)blockIdx.x * blockDim.x + threadIdx.x;
  if (i >= total) return;
  long row = i / G;
  int g = (int)(i % G);
  const float* r = raw + i * 8;
  float mx = r[0];
#pragma unroll
  for (int j = 1; j < 8; ++j) mx = fmaxf(mx, r[j]);
  float e[8], s = 0.f;
#pragma unroll
  for (int j = 0; j < 8; ++j) { e[j] = __expf(r[j] - mx); s += e[j]; }
  float inv = 1.f / s;
#pragma unroll
  for (int j = 0; j < 8; ++j) {
    float w = e[j] * inv;
    dF[row * fs + fo + g * 8 + j] = w;
    if (dB) dB[row * bs + bo + g * 8 + j] = f2bf(w);
  }
}

__global__ void k_lastmix(const float* __restrict__ x, const float* __restrict__ last,
                          u16* __restrict__ dst, long total) {
  long i = (long)blockIdx.x * blockDim.x + threadIdx.x;
  if (i >= total) return;
  int hp = (int)(i % 416);
  long b = i / 416;
  u16 o = 0;
  if (hp < 400) {
    float s = 0.f;
#pragma unroll
    for (int m = 0; m < 8; ++m) s += last[b * 8 + m] * x[(b * 8 + m) * 400 + hp];
    o = f2bf(fmaxf(s, 0.f));
  }
  dst[i] = o;
}

__global__ void k_reparam(const float* __restrict__ enc, const float* __restrict__ eps,
                          u16* __restrict__ dst, long total) {
  long i = (long)blockIdx.x * blockDim.x + threadIdx.x;
  if (i >= total) return;
  int k = (int)(i & 127);
  long b = i >> 7;
  float mu = enc[b * 256 + k], lv = enc[b * 256 + 128 + k];
  dst[i] = f2bf(mu + __expf(0.5f * lv) * eps[i]);
}

// ---------------------------------------------------------------------------
// Host orchestration
// ---------------------------------------------------------------------------
struct GateW {
  const u16 *c0, *c1, *cl, *f0, *f1, *w0, *wa, *wb2, *wl;
  const float *c0b, *c1b, *clb, *f0b, *f1b, *w0b, *wab, *wbb, *wlb;
};
struct Scr {
  u16 *tmp0, *h1, *ebf, *condb, *wbf, *bn1bf, *obf;
  float *e, *wraw, *wf32, *raw8, *last, *mix, *xf, *mean, *rstd;
};

static void gemm(hipStream_t st, const u16* A, int lda, long aB,
                 const u16* BT, int ldb, long bB,
                 const float* bias, long biasB, const float* em, int lde,
                 float* Cf, u16* Cb, int ldc, long cB,
                 int N, int nPad, int K, int batches, int flags) {
  dim3 g(32, (unsigned)((N + 127) / 128), (unsigned)batches), b(256);
  int ks = (K + 31) / 32;
  if (flags == 8)
    k_gemm<8><<<g, b, 0, st>>>(A, lda, aB, BT, ldb, bB, bias, biasB, em, lde, Cf, Cb, ldc, cB, N, nPad, ks);
  else if (flags == 13)
    k_gemm<13><<<g, b, 0, st>>>(A, lda, aB, BT, ldb, bB, bias, biasB, em, lde, Cf, Cb, ldc, cB, N, nPad, ks);
  else
    k_gemm<15><<<g, b, 0, st>>>(A, lda, aB, BT, ldb, bB, bias, biasB, em, lde, Cf, Cb, ldc, cB, N, nPad, ks);
}

#define EW(n) dim3((unsigned)(((n) + 255) / 256)), dim3(256), 0, st

static void cascade(hipStream_t st, const Scr& S, const float* outv, const float* emv,
                    const GateW& G, const u16* modBT, const float* modb,
                    const float* g1, const float* b1, const float* g2, const float* b2,
                    const u16* lastBT, const float* lastb, int lastN,
                    float* lastOut, int lastLdc) {
  const long Bz = 4096;
  // ---- gating ----
  k_mulrelu<<<EW(Bz * 416)>>>(emv, outv, S.tmp0, 400, 416, Bz * 416);
  gemm(st, S.tmp0, 416, 0, G.f0, 416, 0, G.f0b, 0, nullptr, 0, nullptr, S.h1, 512, 0, 512, 512, 400, 1, 13);
  gemm(st, S.h1, 512, 0, G.f1, 512, 0, G.f1b, 0, nullptr, 0, S.e, nullptr, 512, 0, 512, 512, 512, 1, 8);
  k_cvt<<<EW(Bz * 512)>>>(S.e, S.ebf, 512, 512, 1, Bz * 512);
  gemm(st, S.ebf, 512, 0, G.w0, 512, 0, G.w0b, 0, nullptr, 0, S.wraw, nullptr, 64, 0, 64, 64, 512, 1, 8);
  k_softmax8<<<EW(Bz * 8)>>>(S.wraw, 8, Bz * 8, S.wf32, 192, 0, S.wbf, 192, 0);
  gemm(st, S.wbf, 192, 0, G.c0, 64, 0, G.c0b, 0, S.e, 512, nullptr, S.condb, 512, 0, 512, 512, 64, 1, 15);
  gemm(st, S.condb, 512, 0, G.wa, 512, 0, G.wab, 0, nullptr, 0, S.wraw, nullptr, 64, 0, 64, 64, 512, 1, 8);
  k_softmax8<<<EW(Bz * 8)>>>(S.wraw, 8, Bz * 8, S.wf32, 192, 64, S.wbf, 192, 64);
  gemm(st, S.wbf, 192, 0, G.c1, 128, 0, G.c1b, 0, S.e, 512, nullptr, S.condb, 512, 0, 512, 512, 128, 1, 15);
  gemm(st, S.condb, 512, 0, G.wb2, 512, 0, G.wbb, 0, nullptr, 0, S.wraw, nullptr, 64, 0, 64, 64, 512, 1, 8);
  k_softmax8<<<EW(Bz * 8)>>>(S.wraw, 8, Bz * 8, S.wf32, 192, 128, S.wbf, 192, 128);
  gemm(st, S.wbf, 192, 0, G.cl, 192, 0, G.clb, 0, S.e, 512, nullptr, S.condb, 512, 0, 512, 512, 192, 1, 15);
  gemm(st, S.condb, 512, 0, G.wl, 512, 0, G.wlb, 0, nullptr, 0, S.raw8, nullptr, 8, 0, 8, 8, 512, 1, 8);
  k_softmax8<<<EW(Bz)>>>(S.raw8, 1, Bz, S.last, 8, 0, nullptr, 0, 0);

  // ---- module blocks ----
  auto blk = [&](int l) {
    k_colstats<<<dim3(100), dim3(256), 0, st>>>(S.mix, 4096, 3200, S.mean, S.rstd);
    k_bn_bf<<<EW(Bz * 8 * 416)>>>(S.mix, S.mean, S.rstd, g1 + (long)l * 3200,
                                  b1 + (long)l * 3200, S.bn1bf, Bz * 8 * 416);
    gemm(st, S.bn1bf, 3328, 416, modBT + (long)l * 8 * 512 * 416, 416, (long)512 * 416,
         modb + (long)l * 3200, 400, nullptr, 0, S.mix, nullptr, 3200, 400,
         400, 400, 400, 8, 8);
    k_colstats<<<dim3(100), dim3(256), 0, st>>>(S.mix, 4096, 3200, S.mean, S.rstd);
    k_bn_f32<<<EW(Bz * 3200)>>>(S.mix, S.mean, S.rstd, g2 + (long)l * 3200,
                                b2 + (long)l * 3200, S.xf, Bz * 3200);
  };
  k_bcast_relu<<<EW(Bz * 3200)>>>(outv, S.mix, Bz * 3200);
  blk(0);
  for (int i = 0; i < 3; ++i) {
    k_mixture<<<EW(Bz * 400)>>>(S.xf, S.wf32, i * 64, S.mix, Bz * 400);
    blk(i + 1);
  }
  k_lastmix<<<EW(Bz * 416)>>>(S.xf, S.last, S.obf, Bz * 416);
  gemm(st, S.obf, 416, 0, lastBT, 416, 0, lastb, 0, nullptr, 0,
       lastOut, nullptr, lastLdc, 0, lastN, lastN, 400, 1, 8);
}

extern "C" void kernel_launch(void* const* d_in, const int* in_sizes, int n_in,
                              void* d_out, int out_size, void* d_ws, size_t ws_size,
                              hipStream_t st) {
  (void)in_sizes; (void)n_in; (void)out_size; (void)ws_size;
  const long Bz = 4096;
  auto F = [&](int i) -> const float* { return (const float*)d_in[i]; };
  char* wp = (char*)d_ws;
  auto alloc = [&](size_t b) -> void* {
    void* r = wp; wp += (b + 255) & ~(size_t)255; return r;
  };
  auto prepw = [&](int wi, int K, int N) -> const u16* {
    int Kp = (K + 31) / 32 * 32, rows = (N + 127) / 128 * 128;
    long tot = (long)rows * Kp;
    u16* dst = (u16*)alloc((size_t)tot * 2);
    k_prepw<<<EW(tot)>>>(F(wi), dst, K, N, Kp, tot);
    return dst;
  };
  auto prepgate = [&](int gi) -> GateW {
    GateW g;
    g.c0 = prepw(gi + 0, 64, 512);   g.c0b = F(gi + 1);
    g.c1 = prepw(gi + 2, 128, 512);  g.c1b = F(gi + 3);
    g.cl = prepw(gi + 4, 192, 512);  g.clb = F(gi + 5);
    g.f0 = prepw(gi + 6, 400, 512);  g.f0b = F(gi + 7);
    g.f1 = prepw(gi + 8, 512, 512);  g.f1b = F(gi + 9);
    g.w0 = prepw(gi + 10, 512, 64);  g.w0b = F(gi + 11);
    g.wa = prepw(gi + 12, 512, 64);  g.wab = F(gi + 13);
    g.wb2 = prepw(gi + 14, 512, 64); g.wbb = F(gi + 15);
    g.wl = prepw(gi + 16, 512, 8);   g.wlb = F(gi + 17);
    return g;
  };
  auto prepmod = [&](int mi) -> const u16* {
    u16* dst = (u16*)alloc((size_t)32 * 512 * 416 * 2);
    long tot = (long)512 * 416;
    for (int lm = 0; lm < 32; ++lm)
      k_prepw<<<EW(tot)>>>(F(mi) + (long)lm * 400 * 400, dst + (long)lm * 512 * 416,
                           400, 400, 416, tot);
    return dst;
  };

  // weights -> bf16 transposed/padded
  const u16* bt_b0 = prepw(3, 512, 400);
  const u16* bt_b1 = prepw(5, 400, 400);
  const u16* bt_em = prepw(35, 64, 400);
  const u16* bt_de = prepw(7, 128, 400);
  const u16* bt_el = prepw(55, 400, 256);
  const u16* bt_dl = prepw(27, 400, 512);
  GateW Ge = prepgate(37), Gd = prepgate(9);
  const u16* modE = prepmod(57);
  const u16* modD = prepmod(29);

  // scratch
  Scr S;
  S.tmp0 = (u16*)alloc(Bz * 416 * 2);
  S.h1 = (u16*)alloc(Bz * 512 * 2);
  S.ebf = (u16*)alloc(Bz * 512 * 2);
  S.condb = (u16*)alloc(Bz * 512 * 2);
  S.wbf = (u16*)alloc(Bz * 192 * 2);
  S.bn1bf = (u16*)alloc(Bz * 8 * 416 * 2);
  S.obf = (u16*)alloc(Bz * 416 * 2);
  S.e = (float*)alloc(Bz * 512 * 4);
  S.wraw = (float*)alloc(Bz * 64 * 4);
  S.wf32 = (float*)alloc(Bz * 192 * 4);
  S.raw8 = (float*)alloc(Bz * 8 * 4);
  S.last = (float*)alloc(Bz * 8 * 4);
  S.mix = (float*)alloc(Bz * 3200 * 4);
  S.xf = (float*)alloc(Bz * 3200 * 4);
  S.mean = (float*)alloc(3200 * 4);
  S.rstd = (float*)alloc(3200 * 4);
  u16* xbf = (u16*)alloc(Bz * 512 * 2);
  u16* embbf = (u16*)alloc(Bz * 64 * 2);
  u16* hbf = (u16*)alloc(Bz * 416 * 2);
  u16* zbf = (u16*)alloc(Bz * 128 * 2);
  float* base_out = (float*)alloc(Bz * 400 * 4);
  float* em_out = (float*)alloc(Bz * 400 * 4);
  float* de_out = (float*)alloc(Bz * 400 * 4);
  float* enc = (float*)alloc(Bz * 256 * 4);

  // base MLPs
  k_cvt<<<EW(Bz * 512)>>>(F(0), xbf, 512, 512, 0, Bz * 512);
  k_cvt<<<EW(Bz * 64)>>>(F(1), embbf, 64, 64, 0, Bz * 64);
  gemm(st, xbf, 512, 0, bt_b0, 512, 0, F(4), 0, nullptr, 0, nullptr, hbf, 416, 0, 400, 416, 512, 1, 13);
  gemm(st, hbf, 416, 0, bt_b1, 416, 0, F(6), 0, nullptr, 0, base_out, nullptr, 400, 0, 400, 400, 400, 1, 8);
  gemm(st, embbf, 64, 0, bt_em, 64, 0, F(36), 0, nullptr, 0, em_out, nullptr, 400, 0, 400, 400, 64, 1, 8);

  // encoder cascade -> enc (B,256)
  cascade(st, S, base_out, em_out, Ge, modE, F(58), F(61), F(59), F(62), F(60),
          bt_el, F(56), 256, enc, 256);

  // reparameterize + decoder base
  k_reparam<<<EW(Bz * 128)>>>(enc, F(2), zbf, Bz * 128);
  gemm(st, zbf, 128, 0, bt_de, 128, 0, F(8), 0, nullptr, 0, de_out, nullptr, 400, 0, 400, 400, 128, 1, 8);

  // decoder cascade -> recon (B,512) straight into d_out
  cascade(st, S, de_out, em_out, Gd, modD, F(30), F(33), F(31), F(34), F(32),
          bt_dl, F(28), 512, (float*)d_out, 512);
}
